// PositionAwareLayer_24859270709374
// MI455X (gfx1250) — matrix operand order repaired and checked by reading, not compiled, not run on hardware
//
#include <hip/hip_runtime.h>
#include <hip/hip_bf16.h>

// ---------------- problem constants ----------------
#define Bb   2
#define Nn   2048
#define Hh   256
#define NHh  8
#define Dd   32
#define ROWS (Bb * Nn)      // 4096
#define FEAT 312            // H + SPATIAL
#define FPAD 320            // padded feature width (multiple of 32)
#define LNEPS 1e-5f

typedef __attribute__((ext_vector_type(16))) __bf16 v16bf;
typedef __attribute__((ext_vector_type(8)))  __bf16 v8bf;
typedef __attribute__((ext_vector_type(8)))  float  v8f;
typedef int v4i_vs __attribute__((vector_size(16)));   // matches async builtin param

// float -> bf16, round-to-nearest-even
static __device__ __forceinline__ __bf16 f2bf(float x) {
  union { float f; unsigned u; } c; c.f = x;
  unsigned r = (c.u + 0x7FFFu + ((c.u >> 16) & 1u)) >> 16;
  union { unsigned short s; __bf16 b; } o; o.s = (unsigned short)r;
  return o.b;
}

static __device__ __forceinline__ v8f wmma_bf16(v16bf a, v16bf b, v8f c) {
  // (neg_a, A, neg_b, B, c_mod, C, reuse_a, reuse_b)
  return __builtin_amdgcn_wmma_f32_16x16x32_bf16(false, a, false, b, (short)0, c,
                                                 false, false);
}

// Build a 16-element bf16 fragment from two contiguous 8-element (16B) LDS runs.
static __device__ __forceinline__ v16bf frag_pair(const __bf16* p0, const __bf16* p1) {
  v8bf lo = *(const v8bf*)p0;
  v8bf hi = *(const v8bf*)p1;
  v16bf r;
#pragma unroll
  for (int i = 0; i < 8; ++i) { r[i] = lo[i]; r[i + 8] = hi[i]; }
  return r;
}

// ---------------- async global->LDS staging (CDNA5) ----------------
#if __has_builtin(__builtin_amdgcn_global_load_async_to_lds_b128)
#define HAVE_ASYNC 1
#else
#define HAVE_ASYNC 0
#endif

// copy 16 bytes (8 bf16) global -> LDS, tracked by ASYNCcnt
static __device__ __forceinline__ void stage16(const __bf16* g, __bf16* l) {
#if HAVE_ASYNC
  __builtin_amdgcn_global_load_async_to_lds_b128((v4i_vs*)g, (v4i_vs*)l, 0, 0);
#else
  *(v8bf*)l = *(const v8bf*)g;
#endif
}

// wait until at most N async copies are outstanding (per wave, in-order retire)
template <int N>
static __device__ __forceinline__ void async_wait() {
#if HAVE_ASYNC
#if __has_builtin(__builtin_amdgcn_s_wait_asynccnt)
  __builtin_amdgcn_s_wait_asynccnt(N);
#else
  asm volatile("s_wait_asynccnt %0" :: "i"(N) : "memory");
#endif
#endif
}

// =====================================================================
// Kernel 0a: x (f32) -> xbf (bf16), 4096x256
// =====================================================================
__global__ __launch_bounds__(256) void cvt_x_kernel(const float* __restrict__ x,
                                                    __bf16* __restrict__ xbf) {
  int base = blockIdx.x * 1024 + threadIdx.x * 4;
#pragma unroll
  for (int e = 0; e < 4; ++e) xbf[base + e] = f2bf(x[base + e]);
}

// =====================================================================
// Kernel 0b: weights -> bf16, transposed to [col][k] so B staging is
// contiguous 16B chunks. W_out additionally K-padded 312 -> 320 with zeros.
// grid = (80, 4)
// =====================================================================
__global__ __launch_bounds__(256) void cvt_w_kernel(
    const float* __restrict__ Wq, const float* __restrict__ Wk,
    const float* __restrict__ Wv, const float* __restrict__ Wo,
    __bf16* __restrict__ WqT, __bf16* __restrict__ WkT,
    __bf16* __restrict__ WvT, __bf16* __restrict__ WoT) {
  const int y = blockIdx.y;
  const int idx0 = blockIdx.x * 1024 + threadIdx.x * 4;
  if (y < 3) {
    if (idx0 >= Hh * Hh) return;
    const float* W = (y == 0) ? Wq : (y == 1) ? Wk : Wv;
    __bf16* WT     = (y == 0) ? WqT : (y == 1) ? WkT : WvT;
#pragma unroll
    for (int e = 0; e < 4; ++e) {
      int idx = idx0 + e;
      int c = idx >> 8, kk = idx & 255;
      WT[idx] = f2bf(W[(size_t)kk * Hh + c]);
    }
  } else {
#pragma unroll
    for (int e = 0; e < 4; ++e) {
      int idx = idx0 + e;                       // < 256*320 = 81920
      int c = idx / FPAD, kk = idx % FPAD;
      WoT[idx] = f2bf(kk < FEAT ? Wo[(size_t)kk * Hh + c] : 0.0f);
    }
  }
}

// =====================================================================
// Kernel 1: fused QKV projection (bf16 WMMA, double-buffered async staging).
// q,k written bf16 row-major; v pre-transposed vT[bh][d][key].
// grid = (ROWS/128, 3); block = 256 (8 waves).
// =====================================================================
__global__ __launch_bounds__(256) void qkv_kernel(
    const __bf16* __restrict__ xbf,
    const __bf16* __restrict__ WqT, const float* __restrict__ bq,
    const __bf16* __restrict__ WkT, const float* __restrict__ bk,
    const __bf16* __restrict__ WvT, const float* __restrict__ bv,
    __bf16* __restrict__ qbf, __bf16* __restrict__ kbf,
    __bf16* __restrict__ vT) {
  const __bf16* WT; const float* bias;
  if (blockIdx.y == 0)      { WT = WqT; bias = bq; }
  else if (blockIdx.y == 1) { WT = WkT; bias = bk; }
  else                      { WT = WvT; bias = bv; }

  __shared__ alignas(16) __bf16 As[2][128][40];    // ping-pong A tiles
  __shared__ alignas(16) __bf16 BsT[2][256][40];   // ping-pong B^T tiles

  const int tid  = threadIdx.x;
  const int lane = tid & 31, wave = tid >> 5;
  const int g = lane >> 4, n = lane & 15;
  const int row0 = blockIdx.x * 128;
  const int KT = Hh / 32;

  // issue the 6 async chunks (2 A + 4 B) of one K-slice into buffer s
  auto stageAB = [&](int kt, int s) {
    const int k0 = kt * 32;
#pragma unroll
    for (int e = 0; e < 2; ++e) {               // A: 128x32 = 512 x 16B
      int cid = e * 256 + tid;
      int rl = cid >> 2, c8 = (cid & 3) * 8;
      stage16(&xbf[(size_t)(row0 + rl) * Hh + k0 + c8], &As[s][rl][c8]);
    }
#pragma unroll
    for (int e = 0; e < 4; ++e) {               // B^T: 256x32 = 1024 x 16B
      int cid = e * 256 + tid;
      int col = cid >> 2, c8 = (cid & 3) * 8;
      stage16(&WT[(size_t)col * Hh + k0 + c8], &BsT[s][col][c8]);
    }
  };

  v8f acc[16];
#pragma unroll
  for (int t = 0; t < 16; ++t)
#pragma unroll
    for (int r = 0; r < 8; ++r) acc[t][r] = 0.0f;

  stageAB(0, 0);
  for (int kt = 0; kt < KT; ++kt) {
    const int cur = kt & 1;
    if (kt + 1 < KT) { stageAB(kt + 1, cur ^ 1); async_wait<6>(); }
    else             { async_wait<0>(); }
    __syncthreads();

    v16bf af = frag_pair(&As[cur][wave * 16 + n][8 * g],
                         &As[cur][wave * 16 + n][16 + 8 * g]);
#pragma unroll
    for (int nt = 0; nt < 16; ++nt) {
      v16bf wb = frag_pair(&BsT[cur][nt * 16 + n][8 * g],
                           &BsT[cur][nt * 16 + n][16 + 8 * g]);
      acc[nt] = wmma_bf16(af, wb, acc[nt]);
    }
    __syncthreads();
  }

#pragma unroll
  for (int nt = 0; nt < 16; ++nt) {
    int col = nt * 16 + n;
#pragma unroll
    for (int r = 0; r < 8; ++r) {
      int row = row0 + wave * 16 + r + 8 * g;
      float val = acc[nt][r] + bias[col];
      if (blockIdx.y == 2) {                    // v: transposed layout
        int bb = row >> 11, key = row & (Nn - 1);
        int hh = col >> 5,  dd  = col & 31;
        vT[(((size_t)bb * NHh + hh) * Dd + dd) * Nn + key] = f2bf(val);
      } else if (blockIdx.y == 0) {
        qbf[(size_t)row * Hh + col] = f2bf(val);
      } else {
        kbf[(size_t)row * Hh + col] = f2bf(val);
      }
    }
  }
}

// =====================================================================
// Kernel 2: flash attention per (b, h, qtile) + spatial features.
// Double-buffered 32-key tiles; each wave issues exactly ONE async copy
// per tile (waves 0-3 stage K, waves 4-7 stage V^T) -> wait threshold 1.
// grid = (N/128, B*NH); block = 256 (8 waves).
// =====================================================================
__global__ __launch_bounds__(256) void attn_kernel(
    const __bf16* __restrict__ qbf, const __bf16* __restrict__ kbf,
    const __bf16* __restrict__ vT, const float* __restrict__ posCA,
    const float* __restrict__ posCB, const float* __restrict__ frame,
    __bf16* __restrict__ featBf) {
  __shared__ alignas(16) __bf16 Ks[2][32][40];     // keys x dims
  __shared__ alignas(16) __bf16 Vts[2][32][40];    // dims x keys
  __shared__ alignas(16) __bf16 Pws[8][16][40];    // per-wave P tile
  __shared__ float posS[2][96];                    // posCA for 32 keys

  const int tid  = threadIdx.x;
  const int lane = tid & 31, wave = tid >> 5;
  const int g = lane >> 4, n = lane & 15;
  const int qtile = blockIdx.x;
  const int bh = blockIdx.y;
  const int b = bh >> 3, h = bh & 7;
  const size_t bN = (size_t)b * Nn;
  const int row0 = qtile * 128 + wave * 16;        // within-batch base row

  auto stageKV = [&](int kb, int s) {
    if (tid < 128) {
      int key = tid >> 2, c8 = (tid & 3) * 8;
      stage16(&kbf[(bN + kb * 32 + key) * Hh + h * Dd + c8], &Ks[s][key][c8]);
    } else {
      int cid = tid - 128;
      int d = cid >> 2, c8 = (cid & 3) * 8;
      stage16(&vT[((size_t)bh * Dd + d) * Nn + kb * 32 + c8], &Vts[s][d][c8]);
    }
    if (tid < 96) posS[s][tid] = posCA[(bN + kb * 32) * 3 + tid];
  };

  // Q fragment (16 rows x 32 dims), A-layout: K(t) = (t&7) + 8g + 16*(t>>3)
  v16bf qf;
  {
    const int qrow = row0 + n;
#pragma unroll
    for (int t = 0; t < 16; ++t) {
      int kd = (t & 7) + 8 * g + 16 * (t >> 3);
      qf[t] = qbf[(bN + qrow) * Hh + h * Dd + kd];
    }
  }

  v8f o0, o1;
  float mrun[8], lrun[8], ap[8][3];
#pragma unroll
  for (int r = 0; r < 8; ++r) {
    o0[r] = 0.f; o1[r] = 0.f; mrun[r] = -1e30f; lrun[r] = 0.f;
    ap[r][0] = ap[r][1] = ap[r][2] = 0.f;
  }

  stageKV(0, 0);
  for (int kb = 0; kb < Nn / 32; ++kb) {
    const int cur = kb & 1;
    if (kb + 1 < Nn / 32) { stageKV(kb + 1, cur ^ 1); async_wait<1>(); }
    else                  { async_wait<0>(); }
    __syncthreads();

    v8f z;
#pragma unroll
    for (int r = 0; r < 8; ++r) z[r] = 0.f;

    v16bf kb0 = frag_pair(&Ks[cur][n][8 * g],      &Ks[cur][n][16 + 8 * g]);
    v16bf kb1 = frag_pair(&Ks[cur][16 + n][8 * g], &Ks[cur][16 + n][16 + 8 * g]);
    v8f s0 = wmma_bf16(qf, kb0, z);
    v8f s1 = wmma_bf16(qf, kb1, z);

    __bf16* Pw = &Pws[wave][0][0];
#pragma unroll
    for (int r = 0; r < 8; ++r) {   // online softmax per row (r + 8g)
      float t0 = s0[r], t1 = s1[r];
      float mx = fmaxf(t0, t1);
#pragma unroll
      for (int m = 1; m < 16; m <<= 1) mx = fmaxf(mx, __shfl_xor(mx, m, 32));
      float nm   = fmaxf(mrun[r], mx);
      float corr = __expf(mrun[r] - nm);
      float p0 = __expf(t0 - nm);
      float p1 = __expf(t1 - nm);
      float sum = p0 + p1;
#pragma unroll
      for (int m = 1; m < 16; m <<= 1) sum += __shfl_xor(sum, m, 32);
      lrun[r] = lrun[r] * corr + sum;
      mrun[r] = nm;
      o0[r] *= corr; o1[r] *= corr;
#pragma unroll
      for (int c = 0; c < 3; ++c)
        ap[r][c] = ap[r][c] * corr + p0 * posS[cur][n * 3 + c]
                                   + p1 * posS[cur][(n + 16) * 3 + c];
      Pw[(r + 8 * g) * 40 + n]      = f2bf(p0);
      Pw[(r + 8 * g) * 40 + n + 16] = f2bf(p1);
    }

    v16bf pf  = frag_pair(&Pws[wave][n][8 * g], &Pws[wave][n][16 + 8 * g]);
    v16bf vb0 = frag_pair(&Vts[cur][n][8 * g],      &Vts[cur][n][16 + 8 * g]);
    v16bf vb1 = frag_pair(&Vts[cur][16 + n][8 * g], &Vts[cur][16 + n][16 + 8 * g]);
    o0 = wmma_bf16(pf, vb0, o0);
    o1 = wmma_bf16(pf, vb1, o1);
    __syncthreads();
  }

  // finalize: normalize, write feat_node (bf16); reduce alpha@posCA across lanes
#pragma unroll
  for (int r = 0; r < 8; ++r) {
    float inv = 1.0f / lrun[r];
    o0[r] *= inv; o1[r] *= inv;
#pragma unroll
    for (int c = 0; c < 3; ++c) {
#pragma unroll
      for (int m = 1; m < 16; m <<= 1) ap[r][c] += __shfl_xor(ap[r][c], m, 32);
      ap[r][c] *= inv;
    }
    int row = row0 + r + 8 * g;
    size_t fbase = (bN + row) * FPAD;
    featBf[fbase + h * Dd + n]      = f2bf(o0[r]);
    featBf[fbase + h * Dd + 16 + n] = f2bf(o1[r]);
  }

  // spatial features: lane n==r of each half handles row r+8g
#pragma unroll
  for (int r = 0; r < 8; ++r) {
    if (n == r) {
      int row = row0 + r + 8 * g;
      size_t grow = bN + row;
      float apb[3], fp[3];
#pragma unroll
      for (int c = 0; c < 3; ++c) apb[c] = posCB[grow * 3 + c] - ap[r][c];
      float dist = sqrtf(apb[0] * apb[0] + apb[1] * apb[1] + apb[2] * apb[2]);
#pragma unroll
      for (int i = 0; i < 3; ++i)
        fp[i] = frame[grow * 9 + i * 3 + 0] * apb[0] +
                frame[grow * 9 + i * 3 + 1] * apb[1] +
                frame[grow * 9 + i * 3 + 2] * apb[2];
      float fn = sqrtf(fp[0] * fp[0] + fp[1] * fp[1] + fp[2] * fp[2]) + 1e-10f;
      size_t fbase = grow * FPAD;
#pragma unroll
      for (int i = 0; i < 3; ++i) {
        featBf[fbase + 256 + h * 3 + i] = f2bf(fp[i]);       // feat_points
        featBf[fbase + 288 + h * 3 + i] = f2bf(fp[i] / fn);  // feat_direction
      }
      featBf[fbase + 280 + h] = f2bf(dist);                  // feat_distance
    }
  }

  // zero the K-padding columns [312,320) (h==0 blocks cover every row once)
  if (h == 0) {
    int base = qtile * 128;
#pragma unroll
    for (int e = 0; e < 4; ++e) {
      int idx = tid * 4 + e;                 // 0..1023 = 128 rows x 8 cols
      int rl = idx >> 3, c = idx & 7;
      featBf[(bN + base + rl) * FPAD + FEAT + c] = f2bf(0.0f);
    }
  }
}

// =====================================================================
// Kernel 3: y = relu(featAll @ W_out + b_out); LN1; mask; x + y; LN2 -> out
// grid = ROWS/128; block = 256 (8 waves). K = 320, double-buffered staging.
// =====================================================================
__global__ __launch_bounds__(256) void out_kernel(
    const __bf16* __restrict__ featBf, const float* __restrict__ x,
    const __bf16* __restrict__ WoT, const float* __restrict__ bo,
    const float* __restrict__ g1, const float* __restrict__ b1,
    const float* __restrict__ g2, const float* __restrict__ b2,
    const unsigned char* __restrict__ mask, float* __restrict__ out) {
  __shared__ alignas(16) __bf16 As[2][128][40];
  __shared__ alignas(16) __bf16 BsT[2][256][40];

  const int tid  = threadIdx.x;
  const int lane = tid & 31, wave = tid >> 5;
  const int g = lane >> 4, n = lane & 15;
  const int row0 = blockIdx.x * 128;
  const int KT = FPAD / 32;

  auto stageAB = [&](int kt, int s) {
    const int k0 = kt * 32;
#pragma unroll
    for (int e = 0; e < 2; ++e) {               // A: 128x32
      int cid = e * 256 + tid;
      int rl = cid >> 2, c8 = (cid & 3) * 8;
      stage16(&featBf[(size_t)(row0 + rl) * FPAD + k0 + c8], &As[s][rl][c8]);
    }
#pragma unroll
    for (int e = 0; e < 4; ++e) {               // B^T: 256x32
      int cid = e * 256 + tid;
      int col = cid >> 2, c8 = (cid & 3) * 8;
      stage16(&WoT[(size_t)col * FPAD + k0 + c8], &BsT[s][col][c8]);
    }
  };

  v8f acc[16];
#pragma unroll
  for (int t = 0; t < 16; ++t)
#pragma unroll
    for (int r = 0; r < 8; ++r) acc[t][r] = 0.0f;

  stageAB(0, 0);
  for (int kt = 0; kt < KT; ++kt) {
    const int cur = kt & 1;
    if (kt + 1 < KT) { stageAB(kt + 1, cur ^ 1); async_wait<6>(); }
    else             { async_wait<0>(); }
    __syncthreads();

    v16bf af = frag_pair(&As[cur][wave * 16 + n][8 * g],
                         &As[cur][wave * 16 + n][16 + 8 * g]);
#pragma unroll
    for (int nt = 0; nt < 16; ++nt) {
      v16bf wb = frag_pair(&BsT[cur][nt * 16 + n][8 * g],
                           &BsT[cur][nt * 16 + n][16 + 8 * g]);
      acc[nt] = wmma_bf16(af, wb, acc[nt]);
    }
    __syncthreads();
  }

  // fused epilogue: relu -> LN1 -> mask -> +x -> LN2
#pragma unroll
  for (int r = 0; r < 8; ++r) {
    int row = row0 + wave * 16 + r + 8 * g;
    float s1 = 0.f, s2 = 0.f;
#pragma unroll
    for (int nt = 0; nt < 16; ++nt) {
      int col = nt * 16 + n;
      float yv = fmaxf(acc[nt][r] + bo[col], 0.f);
      acc[nt][r] = yv;
      s1 += yv; s2 += yv * yv;
    }
#pragma unroll
    for (int m = 1; m < 16; m <<= 1) {
      s1 += __shfl_xor(s1, m, 32);
      s2 += __shfl_xor(s2, m, 32);
    }
    float mean = s1 * (1.0f / Hh);
    float var  = s2 * (1.0f / Hh) - mean * mean;
    float rstd = rsqrtf(var + LNEPS);
    float mk   = (mask[row] != 0) ? 1.0f : 0.0f;

    float t1 = 0.f, t2 = 0.f;
#pragma unroll
    for (int nt = 0; nt < 16; ++nt) {
      int col = nt * 16 + n;
      float yn = ((acc[nt][r] - mean) * rstd * g1[col] + b1[col]) * mk;
      float zv = x[(size_t)row * Hh + col] + yn;
      acc[nt][r] = zv;
      t1 += zv; t2 += zv * zv;
    }
#pragma unroll
    for (int m = 1; m < 16; m <<= 1) {
      t1 += __shfl_xor(t1, m, 32);
      t2 += __shfl_xor(t2, m, 32);
    }
    float mean2 = t1 * (1.0f / Hh);
    float var2  = t2 * (1.0f / Hh) - mean2 * mean2;
    float rstd2 = rsqrtf(var2 + LNEPS);
#pragma unroll
    for (int nt = 0; nt < 16; ++nt) {
      int col = nt * 16 + n;
      out[(size_t)row * Hh + col] = (acc[nt][r] - mean2) * rstd2 * g2[col] + b2[col];
    }
  }
}

// =====================================================================
extern "C" void kernel_launch(void* const* d_in, const int* in_sizes, int n_in,
                              void* d_out, int out_size, void* d_ws, size_t ws_size,
                              hipStream_t stream) {
  (void)in_sizes; (void)n_in; (void)out_size; (void)ws_size;
  const float* x     = (const float*)d_in[0];
  const float* posCA = (const float*)d_in[1];
  const float* posCB = (const float*)d_in[2];
  const float* frame = (const float*)d_in[3];
  const unsigned char* mask = (const unsigned char*)d_in[4];  // numpy bool
  const float* Wq = (const float*)d_in[5];   const float* bq = (const float*)d_in[6];
  const float* Wk = (const float*)d_in[7];   const float* bk = (const float*)d_in[8];
  const float* Wv = (const float*)d_in[9];   const float* bv = (const float*)d_in[10];
  const float* Wo = (const float*)d_in[11];  const float* bo = (const float*)d_in[12];
  const float* g1 = (const float*)d_in[13];  const float* b1 = (const float*)d_in[14];
  const float* g2 = (const float*)d_in[15];  const float* b2 = (const float*)d_in[16];
  float* out = (float*)d_out;

  // workspace layout (bf16 elements, every region 16B aligned)
  __bf16* w = (__bf16*)d_ws;
  size_t off = 0;
  __bf16* xbf    = w + off; off += (size_t)ROWS * Hh;        // 1,048,576
  __bf16* qbf    = w + off; off += (size_t)ROWS * Hh;
  __bf16* kbf    = w + off; off += (size_t)ROWS * Hh;
  __bf16* vT     = w + off; off += (size_t)ROWS * Hh;        // [bh][d][key]
  __bf16* featBf = w + off; off += (size_t)ROWS * FPAD;      // 1,310,720
  __bf16* WqT    = w + off; off += (size_t)Hh * Hh;
  __bf16* WkT    = w + off; off += (size_t)Hh * Hh;
  __bf16* WvT    = w + off; off += (size_t)Hh * Hh;
  __bf16* WoT    = w + off; off += (size_t)Hh * FPAD;

  cvt_x_kernel<<<dim3(ROWS * Hh / 1024), 256, 0, stream>>>(x, xbf);
  cvt_w_kernel<<<dim3(80, 4), 256, 0, stream>>>(Wq, Wk, Wv, Wo, WqT, WkT, WvT, WoT);
  qkv_kernel<<<dim3(ROWS / 128, 3), 256, 0, stream>>>(xbf, WqT, bq, WkT, bk,
                                                      WvT, bv, qbf, kbf, vT);
  attn_kernel<<<dim3(Nn / 128, Bb * NHh), 256, 0, stream>>>(qbf, kbf, vT, posCA,
                                                            posCB, frame, featBf);
  out_kernel<<<dim3(ROWS / 128), 256, 0, stream>>>(featBf, x, WoT, bo, g1, b1,
                                                   g2, b2, mask, out);
}
